// VSTSRDecoderTransformerBlock_80865644249599
// MI455X (gfx1250) — compile-verified, matrix operand order checked by reading, not run on hardware
//
#include <hip/hip_runtime.h>
#include <hip/hip_bf16.h>

typedef __attribute__((ext_vector_type(16))) _Float16 v16h;
typedef __attribute__((ext_vector_type(8)))  _Float16 v8h;
typedef __attribute__((ext_vector_type(8)))  float    v8f;

#define DEV static __device__ __forceinline__

constexpr int Cdim  = 192;
constexpr int NHEAD = 6;
constexpr int HD    = 32;
constexpr int NTOK  = 128;    // tokens per window (2*8*8)
constexpr int NWIN  = 512;    // total windows
constexpr int MTOT  = 65536;  // total tokens
constexpr int HIDD  = 768;
constexpr int QKVN  = 576;

// ---------------------------------------------------------------------------
// Fragment loader: 16 halves per lane, CDNA5 16-bit A/B layout.
// lane<16: K chunks [0..7] and [16..23]; lane>=16: [8..15] and [24..31].
// p points at (row base + k base); must be 16B aligned.
// ---------------------------------------------------------------------------
DEV v16h load_ab16(const _Float16* p, int lane) {
  p += ((lane >> 4) << 3);
  v8h c0 = *reinterpret_cast<const v8h*>(p);
  v8h c1 = *reinterpret_cast<const v8h*>(p + 16);
  return __builtin_shufflevector(c0, c1, 0,1,2,3,4,5,6,7,8,9,10,11,12,13,14,15);
}

DEV v8f wmma16(v16h a, v16h b, v8f c) {
  return __builtin_amdgcn_wmma_f32_16x16x32_f16(false, a, false, b, (short)0, c,
                                                false, false);
}

DEV float wave_sum32(float v) {
  #pragma unroll
  for (int off = 16; off > 0; off >>= 1) v += __shfl_xor(v, off, 32);
  return v;
}

// ---------------------------------------------------------------------------
// Weight repack fp32 -> fp16.  Layout: [Wq(192x192); Wkv(384x192)] | Wp | Wf1 | Wf2
// ---------------------------------------------------------------------------
__global__ void cvt_weights_kernel(const float* Wq, const float* Wkv, const float* Wp,
                                   const float* Wf1, const float* Wf2, _Float16* W16) {
  int i = blockIdx.x * blockDim.x + threadIdx.x;
  const int S0 = 36864, S1 = 110592, S2 = 147456, S3 = 294912, S4 = 442368;
  if (i >= S4) return;
  float v;
  if      (i < S0) v = Wq [i];
  else if (i < S1) v = Wkv[i - S0];
  else if (i < S2) v = Wp [i - S1];
  else if (i < S3) v = Wf1[i - S2];
  else             v = Wf2[i - S3];
  W16[i] = (_Float16)v;
}

// ---------------------------------------------------------------------------
// Masked-bias table: tbl[head][cls][t1][t2] = rel_pos_bias + shift_mask.
// cls = (dw==3)*4 | (hw==7)*2 | (ww==7): only the wrapped windows mix regions.
// ---------------------------------------------------------------------------
__global__ void bias_table_kernel(const float* __restrict__ rpb,
                                  float* __restrict__ tbl) {
  int idx = blockIdx.x * 256 + threadIdx.x;          // 6*8*128*128 = 786432
  if (idx >= NHEAD * 8 * NTOK * NTOK) return;
  int t2   = idx & 127;
  int t1   = (idx >> 7) & 127;
  int cls  = (idx >> 14) & 7;
  int head = idx >> 17;
  int td1 = t1 >> 6, th1 = (t1 >> 3) & 7, tw1 = t1 & 7;
  int td2 = t2 >> 6, th2 = (t2 >> 3) & 7, tw2 = t2 & 7;
  int rel = (td1 - td2 + 1) * 225 + (th1 - th2 + 7) * 15 + (tw1 - tw2 + 7);
  float bias = rpb[rel * NHEAD + head];
  int dl = (cls >> 2) & 1, hl = (cls >> 1) & 1, wl = cls & 1;
  // region ids within this window class (interior window -> region 0 per axis)
  int r1 = (dl ? (1 + td1) : 0) * 9 + (hl ? ((th1 >= 4) ? 2 : 1) : 0) * 3
         + (wl ? ((tw1 >= 4) ? 2 : 1) : 0);
  int r2 = (dl ? (1 + td2) : 0) * 9 + (hl ? ((th2 >= 4) ? 2 : 1) : 0) * 3
         + (wl ? ((tw2 >= 4) ? 2 : 1) : 0);
  tbl[idx] = bias + ((r1 != r2) ? -100.0f : 0.0f);
}

// ---------------------------------------------------------------------------
// LayerNorm, one wave per token (6 elems/lane, 32 lanes = 192 channels).
// GATHER=true: token index is window-order; source gathered through shift-roll.
// ---------------------------------------------------------------------------
template<bool GATHER>
__global__ __launch_bounds__(256) void ln_kernel(const float* __restrict__ x,
                                                 const float* __restrict__ g,
                                                 const float* __restrict__ b,
                                                 _Float16* __restrict__ o) {
  int lane  = threadIdx.x & 31;
  int wave  = threadIdx.x >> 5;
  int token = blockIdx.x * 8 + wave;          // [0, 65536)
  size_t src;
  if constexpr (GATHER) {
    int win = token >> 7, t = token & 127;
    int bb = win >> 8, rw = win & 255;
    int dw = rw >> 6, hw = (rw >> 3) & 7, ww = rw & 7;
    int td = t >> 6,  th = (t >> 3) & 7,  tw = t & 7;
    int g0 = (dw * 2 + td + 1) & 7;           // roll by shift (1,4,4)
    int g1 = (hw * 8 + th + 4) & 63;
    int g2 = (ww * 8 + tw + 4) & 63;
    src = (((size_t)bb * 8 + g0) * 64 + g1) * 64 + g2;
  } else {
    src = (size_t)token;
  }
  const float* xp = x + src * Cdim;
  float v[6]; float s = 0.f;
  #pragma unroll
  for (int i = 0; i < 6; i++) { v[i] = xp[lane + 32 * i]; s += v[i]; }
  float mean = wave_sum32(s) * (1.0f / 192.0f);
  float q = 0.f;
  #pragma unroll
  for (int i = 0; i < 6; i++) { float d = v[i] - mean; q += d * d; }
  float rstd = rsqrtf(wave_sum32(q) * (1.0f / 192.0f) + 1e-5f);
  _Float16* op = o + (size_t)token * Cdim;
  #pragma unroll
  for (int i = 0; i < 6; i++) {
    int c = lane + 32 * i;
    op[c] = (_Float16)((v[i] - mean) * rstd * g[c] + b[c]);
  }
}

// ---------------------------------------------------------------------------
// WMMA GEMM: C[MTOT][N] = A[MTOT][K](f16) * W[N][K]^T(f16), wave = 32x64 tile.
// N, K compile-time: k-loop fully unrolls, grid divides exactly (no branch).
// ---------------------------------------------------------------------------
enum { EPI_QKV = 0, EPI_PROJ = 1, EPI_GELU = 2, EPI_FINAL = 3 };

template<int EPI, int N, int K>
__global__ __launch_bounds__(256) void gemm_kernel(const _Float16* __restrict__ A,
                                                   const _Float16* __restrict__ W,
                                                   const float* __restrict__ bias,
                                                   const float* __restrict__ resid,
                                                   _Float16* __restrict__ outH,
                                                   float* __restrict__ outF) {
  constexpr int NG = N >> 6;
  int lane = threadIdx.x & 31;
  int gw   = blockIdx.x * 8 + (threadIdx.x >> 5);  // wave-uniform, always valid
  int mtile = gw / NG, ngrp = gw - mtile * NG;
  int m0 = mtile << 5, n0 = ngrp << 6;

  v8f acc[2][4] = {};
  const _Float16* arow0 = A + (size_t)(m0 + (lane & 15)) * K;
  const _Float16* arow1 = arow0 + (size_t)16 * K;
  const _Float16* wrow  = W + (size_t)(n0 + (lane & 15)) * K;
  #pragma unroll
  for (int k0 = 0; k0 < K; k0 += 32) {
    if constexpr (true) {
      if (k0 + 32 < K) {                     // prefetch next k-block (uniform)
        __builtin_prefetch(arow0 + k0 + 32, 0, 3);
        __builtin_prefetch(arow1 + k0 + 32, 0, 3);
        #pragma unroll
        for (int j = 0; j < 4; j++)
          __builtin_prefetch(wrow + (size_t)(j * 16) * K + k0 + 32, 0, 3);
      }
    }
    v16h a0 = load_ab16(arow0 + k0, lane);
    v16h a1 = load_ab16(arow1 + k0, lane);
    #pragma unroll
    for (int j = 0; j < 4; j++) {
      v16h bf = load_ab16(wrow + (size_t)(j * 16) * K + k0, lane);
      acc[0][j] = wmma16(a0, bf, acc[0][j]);
      acc[1][j] = wmma16(a1, bf, acc[1][j]);
    }
  }

  int rbase = (lane >> 4) << 3;    // C layout: vgpr r -> row r (+8 for hi lanes)
  int colb  = lane & 15;
  #pragma unroll
  for (int h2 = 0; h2 < 2; h2++) {
    #pragma unroll
    for (int j = 0; j < 4; j++) {
      #pragma unroll
      for (int r = 0; r < 8; r++) {
        int row = m0 + h2 * 16 + rbase + r;
        int col = n0 + j * 16 + colb;
        float v = acc[h2][j][r];
        if constexpr (EPI == EPI_QKV) {
          // fold 1/sqrt(hd) into Q columns (col < 192)
          float s = (col < Cdim) ? 0.17677669529663687f : 1.0f;
          outH[(size_t)row * N + col] = (_Float16)(v * s);
        } else if constexpr (EPI == EPI_GELU) {
          float t = v + bias[col];
          float gl = 0.5f * t * (1.0f + erff(t * 0.70710678118654752f));
          outH[(size_t)row * N + col] = (_Float16)gl;
        } else if constexpr (EPI == EPI_PROJ) {
          // window-reverse + reverse shift-roll scatter + residual
          int win = row >> 7, t = row & 127;
          int bb = win >> 8, rw2 = win & 255;
          int dw = rw2 >> 6, hw = (rw2 >> 3) & 7, ww = rw2 & 7;
          int td = t >> 6,  th = (t >> 3) & 7,  tw = t & 7;
          int g0 = (dw * 2 + td + 1) & 7;
          int g1 = (hw * 8 + th + 4) & 63;
          int g2 = (ww * 8 + tw + 4) & 63;
          size_t dst = ((((size_t)bb * 8 + g0) * 64 + g1) * 64 + g2) * Cdim + col;
          outF[dst] = resid[dst] + v + bias[col];
        } else { // EPI_FINAL
          size_t idx = (size_t)row * Cdim + col;
          outF[idx] = resid[idx] + v + bias[col];
        }
      }
    }
  }
}

// ---------------------------------------------------------------------------
// Windowed attention: block = (window, head), 8 waves x 16 query rows.
// Q already carries the 1/sqrt(hd) scale; bias+mask comes from the table.
// ---------------------------------------------------------------------------
__global__ __launch_bounds__(256) void attn_kernel(const _Float16* __restrict__ qkv,
                                                   const float* __restrict__ btbl,
                                                   _Float16* __restrict__ attn_out) {
  __shared__ __align__(16) _Float16 sV[32][136];       // V^T : [hd][key]
  __shared__ __align__(16) _Float16 sP[8][16][136];    // probs per wave

  int lane = threadIdx.x & 31, wave = threadIdx.x >> 5;
  int head = blockIdx.x % NHEAD;
  int win  = blockIdx.x / NHEAD;
  int rw   = win & 255;
  int dw = rw >> 6, hw = (rw >> 3) & 7, ww = rw & 7;
  int cls = ((dw == 3) ? 4 : 0) | ((hw == 7) ? 2 : 0) | ((ww == 7) ? 1 : 0);
  size_t wbase = (size_t)win * NTOK * QKVN;

  // Stage V transposed into LDS (coalesced rows -> transposed columns)
  for (int i = threadIdx.x; i < NTOK * HD; i += 256) {
    int key = i >> 5, c = i & 31;
    sV[c][key] = qkv[wbase + (size_t)key * QKVN + 2 * Cdim + head * HD + c];
  }
  __syncthreads();

  int qr0 = wave << 4;
  int rbase = (lane >> 4) << 3;
  int colb  = lane & 15;
  v16h qa = load_ab16(qkv + wbase + (size_t)(qr0 + (lane & 15)) * QKVN + head * HD, lane);

  // scores = Q K^T  (8 key tiles, K-dim = hd = 32 in one WMMA each)
  v8f sc[8];
  #pragma unroll
  for (int kt = 0; kt < 8; kt++) {
    v16h kb = load_ab16(qkv + wbase + (size_t)(kt * 16 + (lane & 15)) * QKVN
                            + Cdim + head * HD, lane);
    v8f z = {};
    sc[kt] = wmma16(qa, kb, z);
  }

  // add precomputed bias+mask: one coalesced load per element
  const float* bt = btbl + ((size_t)head * 8 + cls) * NTOK * NTOK;
  #pragma unroll
  for (int r = 0; r < 8; r++) {
    const float* btr = bt + (qr0 + rbase + r) * NTOK + colb;
    #pragma unroll
    for (int kt = 0; kt < 8; kt++) sc[kt][r] += btr[kt * 16];
  }

  // row softmax: 16-lane xor reductions (two rows per vgpr, one per lane-half)
  #pragma unroll
  for (int r = 0; r < 8; r++) {
    float mx = sc[0][r];
    #pragma unroll
    for (int kt = 1; kt < 8; kt++) mx = fmaxf(mx, sc[kt][r]);
    #pragma unroll
    for (int off = 1; off < 16; off <<= 1) mx = fmaxf(mx, __shfl_xor(mx, off, 32));
    float sum = 0.f;
    #pragma unroll
    for (int kt = 0; kt < 8; kt++) {
      float e = __expf(sc[kt][r] - mx); sc[kt][r] = e; sum += e;
    }
    #pragma unroll
    for (int off = 1; off < 16; off <<= 1) sum += __shfl_xor(sum, off, 32);
    float inv = 1.0f / sum;
    int rl = rbase + r;
    #pragma unroll
    for (int kt = 0; kt < 8; kt++)
      sP[wave][rl][kt * 16 + colb] = (_Float16)(sc[kt][r] * inv);
  }
  __syncthreads();

  // out = P V : A-frags from LDS probs, B-frags from transposed V
  v8f o0 = {}, o1 = {};
  #pragma unroll
  for (int kc = 0; kc < 4; kc++) {
    v16h pa  = load_ab16(&sP[wave][lane & 15][kc * 32], lane);
    v16h vb0 = load_ab16(&sV[colb][kc * 32], lane);
    v16h vb1 = load_ab16(&sV[colb + 16][kc * 32], lane);
    o0 = wmma16(pa, vb0, o0);
    o1 = wmma16(pa, vb1, o1);
  }
  #pragma unroll
  for (int r = 0; r < 8; r++) {
    int t = qr0 + rbase + r;
    size_t off = ((size_t)win * NTOK + t) * Cdim + head * HD;
    attn_out[off + colb]      = (_Float16)o0[r];
    attn_out[off + 16 + colb] = (_Float16)o1[r];
  }
}

// ---------------------------------------------------------------------------
extern "C" void kernel_launch(void* const* d_in, const int* in_sizes, int n_in,
                              void* d_out, int out_size, void* d_ws, size_t ws_size,
                              hipStream_t stream) {
  const float* x   = (const float*)d_in[0];
  const float* g1  = (const float*)d_in[1];
  const float* b1  = (const float*)d_in[2];
  const float* Wq  = (const float*)d_in[3];
  const float* Wkv = (const float*)d_in[4];
  const float* rpb = (const float*)d_in[5];
  const float* Wp  = (const float*)d_in[6];
  const float* bp  = (const float*)d_in[7];
  const float* g2  = (const float*)d_in[8];
  const float* b2  = (const float*)d_in[9];
  const float* Wf1 = (const float*)d_in[10];
  const float* bf1 = (const float*)d_in[11];
  const float* Wf2 = (const float*)d_in[12];
  const float* bf2 = (const float*)d_in[13];
  float* out = (float*)d_out;

  // workspace (bytes): weights | xw/h | qkv(+attn) [reused as f1] | x2 | bias tbl
  char* ws = (char*)d_ws;
  _Float16* W16      = (_Float16*)ws;                                      //  0.9 MB
  _Float16* xw       = (_Float16*)(ws + 1048576u);                         // 25.2 MB
  _Float16* qkv      = (_Float16*)(ws + 1048576u + 25165824u);             // 75.5 MB
  _Float16* attn_out = (_Float16*)(ws + 1048576u + 25165824u + 75497472u); // 25.2 MB
  _Float16* f1       = qkv;   // reuse qkv+attn regions (exactly 100663296 B)
  _Float16* h        = xw;    // reuse xw region
  float*    x2       = (float*)(ws + 1048576u + 25165824u + 100663296u);   // 50.3 MB
  float*    btbl     = (float*)(ws + 1048576u + 25165824u + 100663296u + 50331648u); // 3 MB

  cvt_weights_kernel<<<(442368 + 255) / 256, 256, 0, stream>>>(Wq, Wkv, Wp, Wf1, Wf2, W16);
  bias_table_kernel<<<(NHEAD * 8 * NTOK * NTOK + 255) / 256, 256, 0, stream>>>(rpb, btbl);
  ln_kernel<true><<<MTOT / 8, 256, 0, stream>>>(x, g1, b1, xw);
  gemm_kernel<EPI_QKV, QKVN, Cdim><<<(2048 * 9) / 8, 256, 0, stream>>>(
      xw, W16, nullptr, nullptr, qkv, nullptr);
  attn_kernel<<<NWIN * NHEAD, 256, 0, stream>>>(qkv, btbl, attn_out);
  gemm_kernel<EPI_PROJ, Cdim, Cdim><<<(2048 * 3) / 8, 256, 0, stream>>>(
      attn_out, W16 + 110592, bp, x, nullptr, x2);
  ln_kernel<false><<<MTOT / 8, 256, 0, stream>>>(x2, g2, b2, h);
  gemm_kernel<EPI_GELU, HIDD, Cdim><<<(2048 * 12) / 8, 256, 0, stream>>>(
      h, W16 + 147456, bf1, nullptr, f1, nullptr);
  gemm_kernel<EPI_FINAL, Cdim, HIDD><<<(2048 * 3) / 8, 256, 0, stream>>>(
      f1, W16 + 294912, bf2, x2, nullptr, out);
}
